// SmallMLP_71167608094757
// MI455X (gfx1250) — matrix operand order repaired and checked by reading, not compiled, not run on hardware
//
#include <hip/hip_runtime.h>
#include <math.h>

// ---------------------------------------------------------------------------
// SmallMLP on MI455X (gfx1250, wave32, WMMA)
//
//   h      = ternarize( x[65536,784] @ tern(w1)[320,784]^T + b1 )
//   logits = h @ tern(w2)[10,320]^T + b2 ;  out = log_softmax(logits)
//
// Layer-1 GEMM (32.9 GFLOP) dominates -> v_wmma_f32_16x16x32_f16 with an
// f32 = hi_f16 + lo_f16 split of x (weights are exactly {-1,0,1} in f16),
// f32 accumulation. x (205 MB) streams from HBM once; w1 (1 MB) lives in L2.
// B fragments are register double-buffered so ds_load latency hides under
// the WMMA pipe; next K-chunk of x is prefetched (global_prefetch_b8).
// ---------------------------------------------------------------------------

typedef _Float16 half8  __attribute__((ext_vector_type(8)));
typedef _Float16 half16 __attribute__((ext_vector_type(16)));
typedef float    float8 __attribute__((ext_vector_type(8)));

#define TERN_TH 0.001f

// ---- LDS union layout (bytes) ---------------------------------------------
// Phase 1 (K loop):
//   sAhi : [128][40] f16   @ 0       (10240)   x hi plane, stride 40 halves -> bank-conflict-free b128
//   sAlo : [128][40] f16   @ 10240   (10240)   x lo plane
//   sW   : [320][40] f16   @ 20480   (25600)   ternarized w1 chunk
// Phase 2 (after barrier, regions reused):
//   hT   : [128][324] i8   @ 0       (41472)   ternarized hidden acts (stride 324 -> conflict-free)
//   q2s  : [10][320] f32   @ 41472   (12800)   ternarized w2
static constexpr int SMEM_BYTES = 55296;

__device__ __forceinline__ half16 load_frag16(const _Float16* p0, const _Float16* p1)
{
    half8 a = *(const half8*)p0;
    half8 b = *(const half8*)p1;
    return __builtin_shufflevector(a, b, 0,1,2,3,4,5,6,7,8,9,10,11,12,13,14,15);
}

__global__ __launch_bounds__(256)
void smallmlp_wmma_kernel(const float* __restrict__ x,
                          const float* __restrict__ w1,
                          const float* __restrict__ b1,
                          const float* __restrict__ w2,
                          const float* __restrict__ b2,
                          float* __restrict__ out)
{
    __shared__ __align__(16) unsigned char smem[SMEM_BYTES];
    _Float16* sAhi = (_Float16*)(smem);
    _Float16* sAlo = (_Float16*)(smem + 10240);
    _Float16* sW   = (_Float16*)(smem + 20480);

    const int tid  = threadIdx.x;
    const int wave = tid >> 5;        // 0..7
    const int lane = tid & 31;
    const int l16  = lane & 15;
    const int g    = lane >> 4;       // lane half: 0 or 1
    const int wgM  = blockIdx.x * 128;

    float8 acc[20];                   // 20 N-tiles of 16x16 f32 = 160 VGPRs
#pragma unroll
    for (int i = 0; i < 20; ++i) acc[i] = (float8)0.0f;

    // ---------------- K loop: 784 = 24*32 + 16 (last chunk zero-padded) ----
    for (int kc = 0; kc < 25; ++kc) {
        const int k0 = kc * 32;

        // x chunk -> hi/lo f16 planes (coalesced: 32 lanes cover one row's 32 k)
#pragma unroll
        for (int i = 0; i < 16; ++i) {
            int e  = i * 256 + tid;           // 0..4095 = 128 rows x 32 k
            int r  = e >> 5;
            int kk = e & 31;                  // == lane
            int k  = k0 + kk;
            const float* xrow = x + (size_t)(wgM + r) * 784;
            float v = (k < 784) ? xrow[k] : 0.0f;
            // prefetch next K-chunk: one 128B line per row, issued by lane 0
            if (kk == 0 && k0 + 32 < 784)
                __builtin_prefetch(xrow + k0 + 32, 0, 1);
            _Float16 hi = (_Float16)v;
            _Float16 lo = (_Float16)(v - (float)hi);
            sAhi[r * 40 + kk] = hi;
            sAlo[r * 40 + kk] = lo;
        }
        // w1 chunk -> ternarized f16 (clamp is subsumed by ternarize, TH<1)
#pragma unroll 8
        for (int i = 0; i < 40; ++i) {
            int e  = i * 256 + tid;           // 0..10239 = 320 rows x 32 k
            int n  = e >> 5;
            int kk = e & 31;
            int k  = k0 + kk;
            float v = (k < 784) ? w1[(size_t)n * 784 + k] : 0.0f;
            _Float16 q = (v > TERN_TH) ? (_Float16)1.0f
                        : ((v < -TERN_TH) ? (_Float16)-1.0f : (_Float16)0.0f);
            sW[n * 40 + kk] = q;
        }
        __syncthreads();

        // A fragments (16x32 f16): lane(l16=M, g): K = 16g + e
        const _Float16* aRowH = sAhi + (wave * 16 + l16) * 40;
        const _Float16* aRowL = sAlo + (wave * 16 + l16) * 40;
        half16 aH = load_frag16(aRowH + 16 * g, aRowH + 16 * g + 8);
        half16 aL = load_frag16(aRowL + 16 * g, aRowL + 16 * g + 8);

        // B fragment (32x16): lane = column n = w1 row; K = 16g + e.
        // Register double-buffer: issue tile nt+1's ds_loads before tile nt's
        // WMMAs so LDS latency hides under the matrix pipe.
        const _Float16* bBase = sW + l16 * 40 + 16 * g;
        half16 bCur = load_frag16(bBase, bBase + 8);
#pragma unroll
        for (int nt = 0; nt < 20; ++nt) {
            half16 bNext;
            if (nt < 19) {
                const _Float16* bp = bBase + (nt + 1) * 16 * 40;
                bNext = load_frag16(bp, bp + 8);
            }
            acc[nt] = __builtin_amdgcn_wmma_f32_16x16x32_f16(
                          false, aH, false, bCur, (short)0, acc[nt], false, false);
            acc[nt] = __builtin_amdgcn_wmma_f32_16x16x32_f16(
                          false, aL, false, bCur, (short)0, acc[nt], false, false);
            if (nt < 19) bCur = bNext;
        }
        __syncthreads();   // protects next chunk's LDS overwrite (and phase 2)
    }

    // ---------------- Phase 2: bias + ternarize h -> LDS (int8) ------------
    signed char* hT  = (signed char*)(smem);           // [128][324]
    float*       q2s = (float*)(smem + 41472);         // [10][320]

#pragma unroll
    for (int nt = 0; nt < 20; ++nt) {
        int n = nt * 16 + l16;
        float bias = b1[n];
#pragma unroll
        for (int r = 0; r < 8; ++r) {                  // C layout: M = r + 8g, N = l16
            int mloc = wave * 16 + g * 8 + r;
            float h  = acc[nt][r] + bias;
            signed char t = (h > TERN_TH) ? 1 : ((h < -TERN_TH) ? -1 : 0);
            hT[mloc * 324 + n] = t;
        }
    }
    // ternarized w2 -> LDS as f32
    for (int i = 0; i < 13; ++i) {
        int e = i * 256 + tid;
        if (e < 3200) {
            float v = w2[e];
            q2s[e] = (v > TERN_TH) ? 1.0f : ((v < -TERN_TH) ? -1.0f : 0.0f);
        }
    }
    __syncthreads();

    // ---------------- Phase 3: layer 2 (N=10) + log_softmax ----------------
    if (tid < 128) {
        float lg[10];
#pragma unroll
        for (int o = 0; o < 10; ++o) lg[o] = b2[o];
        const signed char* hrow = hT + tid * 324;
        for (int n = 0; n < 320; ++n) {
            float s = (float)hrow[n];
#pragma unroll
            for (int o = 0; o < 10; ++o) lg[o] += s * q2s[o * 320 + n];
        }
        float mx = lg[0];
#pragma unroll
        for (int o = 1; o < 10; ++o) mx = fmaxf(mx, lg[o]);
        float sum = 0.0f;
#pragma unroll
        for (int o = 0; o < 10; ++o) sum += expf(lg[o] - mx);
        float lse = logf(sum) + mx;
        float* orow = out + (size_t)(wgM + tid) * 10;
#pragma unroll
        for (int o = 0; o < 10; ++o) orow[o] = lg[o] - lse;
    }
}

extern "C" void kernel_launch(void* const* d_in, const int* in_sizes, int n_in,
                              void* d_out, int out_size, void* d_ws, size_t ws_size,
                              hipStream_t stream)
{
    (void)in_sizes; (void)n_in; (void)d_ws; (void)ws_size; (void)out_size;
    const float* x  = (const float*)d_in[0];   // [65536, 784]
    const float* w1 = (const float*)d_in[1];   // [320, 784]
    const float* b1 = (const float*)d_in[2];   // [320]
    const float* w2 = (const float*)d_in[3];   // [10, 320]
    const float* b2 = (const float*)d_in[4];   // [10]
    float* out = (float*)d_out;                // [65536, 10]

    dim3 grid(65536 / 128);                    // 512 workgroups, 128 rows each
    dim3 block(256);                           // 8 wave32 waves
    smallmlp_wmma_kernel<<<grid, block, 0, stream>>>(x, w1, b1, w2, b2, out);
}
